// AtacformerPairwiseInteractionHead_79044578116265
// MI455X (gfx1250) — matrix operand order repaired and checked by reading, not compiled, not run on hardware
//
#include <hip/hip_runtime.h>

typedef __attribute__((ext_vector_type(16))) _Float16 v16h;
typedef __attribute__((ext_vector_type(8)))  _Float16 v8h;
typedef __attribute__((ext_vector_type(8)))  float    v8f;
typedef __attribute__((ext_vector_type(4)))  unsigned int v4u;

#define B_ 4
#define N_ 256
#define D_ 512

// workspace layout (bytes)
static constexpr size_t EH_OFF    = 0;                          // f16 [B][N][D]   = 1 MB
static constexpr size_t APROJ_OFF = (size_t)1 << 20;            // f32 [B][N][D]   = 2 MB
static constexpr size_t CPROJ_OFF = (size_t)3 << 20;            // f32 [B][N][D]   = 2 MB
static constexpr size_t WCF_OFF   = (size_t)5 << 20;            // f16 fragments   = 512 KB
static constexpr size_t WDF_OFF   = ((size_t)5 << 20) + (size_t)D_ * D_ * 2;
// total ~6 MB

// ---------------------------------------------------------------------------
// Kernel 0: embeddings f32 -> f16
// ---------------------------------------------------------------------------
__global__ __launch_bounds__(256) void eh_kernel(const float* __restrict__ e,
                                                 _Float16* __restrict__ eh) {
  size_t gid = (size_t)blockIdx.x * 256 + threadIdx.x;   // < B*N*D = 2^19
  eh[gid] = (_Float16)e[gid];
}

// ---------------------------------------------------------------------------
// Kernel 1: aproj = e @ Wa, cproj = e @ Wb   (f32, tiny fraction of FLOPs)
// ---------------------------------------------------------------------------
__global__ __launch_bounds__(256) void proj_kernel(const float* __restrict__ e,
                                                   const float* __restrict__ W1,
                                                   float* __restrict__ aproj,
                                                   float* __restrict__ cproj) {
  size_t gid = (size_t)blockIdx.x * 256 + threadIdx.x;   // < 2*B*N*D = 2^20
  int    k   = (int)(gid & (D_ - 1));                    // bits [8:0]
  size_t bn  = (gid >> 9) & (B_ * N_ - 1);               // bits [18:9]
  int    sel = (int)(gid >> 19);                         // 0 = Wa, 1 = Wb
  const float* erow = e + bn * D_;
  const float* w    = W1 + (size_t)sel * D_ * D_;        // Wa rows 0..511, Wb rows 512..1023
  float s = 0.f;
  for (int d = 0; d < D_; ++d) s += erow[d] * w[(size_t)d * D_ + k];
  (sel ? cproj : aproj)[bn * D_ + k] = s;
}

// ---------------------------------------------------------------------------
// Kernel 2: pre-swizzle Wc/Wd (f32 -> f16) into WMMA B-fragment order.
// B 32x16 f16 layout (wave32): lane l holds column N=l&15, halves h=0..15 map
// to K = (h&7) + (h>=8)*16 + (l>=16)*8. Fragment (dt,kt) = 512 halves, lane-major.
// ---------------------------------------------------------------------------
__global__ __launch_bounds__(256) void pack_kernel(const float* __restrict__ W1,
                                                   _Float16* __restrict__ wcf,
                                                   _Float16* __restrict__ wdf) {
  size_t gid = (size_t)blockIdx.x * 256 + threadIdx.x;   // < 2*16*32*512 = 2^19
  int h    = (int)(gid & 15);
  int lane = (int)((gid >> 4) & 31);
  int kt   = (int)((gid >> 9) & 31);
  int dt   = (int)((gid >> 14) & 15);
  int mat  = (int)((gid >> 18) & 1);                     // 0 = Wc, 1 = Wd
  int kb   = (h & 7) + ((h >> 3) << 4) + ((lane >> 4) << 3); // K within 32-chunk
  int d    = dt * 32 + kb;
  int kk   = kt * 16 + (lane & 15);
  float v  = W1[((size_t)(1024 + mat * 512 + d)) * D_ + kk];
  _Float16* dst = mat ? wdf : wcf;
  dst[((size_t)(dt * 32 + kt)) * 512 + lane * 16 + h] = (_Float16)v;
}

// ---------------------------------------------------------------------------
// Main kernel: one WG per (b, 16 i's, 16 j's); 8 waves; wave w owns the
// 64-wide k slice [w*64, w*64+64) = 4 k-tiles of 16. Fused epilogue.
// ---------------------------------------------------------------------------
__global__ __launch_bounds__(256) void pairhead_main(
    const _Float16* __restrict__ eh, const float* __restrict__ aproj,
    const float* __restrict__ cproj, const _Float16* __restrict__ wcf,
    const _Float16* __restrict__ wdf, const float* __restrict__ b1,
    const float* __restrict__ W2, const float* __restrict__ b2,
    float* __restrict__ out) {
  __shared__ _Float16 s_e[2 * 16 * D_];   // e_i tile | e_j tile (f16, 32 KB)
  __shared__ float s_scores[256];         // 16x16 partial scores

  const int tid  = threadIdx.x;
  const int lane = tid & 31;
  const int wave = tid >> 5;              // 0..7 -> k slice
  const int blk  = blockIdx.x;            // b*256 + it*16 + jt
  const int b    = blk >> 8;
  const int it   = (blk >> 4) & 15;
  const int jt   = blk & 15;
  const int i0   = it * 16, j0 = jt * 16;

  _Float16* s_ei = s_e;
  _Float16* s_ej = s_e + 16 * D_;

  s_scores[tid] = 0.f;

  // cooperative stage of 16+16 embedding rows (f16), 8-half vectors
  for (int c = tid; c < 16 * D_ / 8; c += 256) {
    int row = c >> 6;
    int col = (c & 63) * 8;
    *reinterpret_cast<v8h*>(&s_ei[row * D_ + col]) =
        *reinterpret_cast<const v8h*>(&eh[((size_t)(b * N_) + i0 + row) * D_ + col]);
    *reinterpret_cast<v8h*>(&s_ej[row * D_ + col]) =
        *reinterpret_cast<const v8h*>(&eh[((size_t)(b * N_) + j0 + row) * D_ + col]);
  }
  __syncthreads();

  // Hoisted epilogue constants: the k columns this wave owns, and the cproj
  // tile values -- none of these depend on il, so load them ONCE per WG.
  // C layout: VGPR r, lane l -> M = r + (l>>4)*8 (=j_local), N = l&15 (=k).
  const int kbase = wave * 64 + (lane & 15);
  float b1k[4], w2k[4], cjk[4][8];
#pragma unroll
  for (int t = 0; t < 4; ++t) {
    const int k = kbase + t * 16;
    b1k[t] = b1[k];
    w2k[t] = W2[k];
#pragma unroll
    for (int r = 0; r < 8; ++r) {
      const int jl = r + (lane >> 4) * 8;
      cjk[t][r] = cproj[((size_t)(b * N_) + j0 + jl) * D_ + k];
    }
  }

  // A-fragment lane geometry (16-bit A 16x32):
  //   row M = lane&15; K = (h&7) + (h>=8)*16 + (lane>=16)*8
  const int m   = lane & 15;
  const int off = (lane >> 4) * 8;

  const float* aprow_base = aproj + ((size_t)(b * N_) + i0) * D_;

  for (int il = 0; il < 16; ++il) {
    v8f acc[4];
    acc[0] = {}; acc[1] = {}; acc[2] = {}; acc[3] = {};
    const _Float16* eirow = &s_ei[il * D_];
    const _Float16* ejrow = &s_ej[m * D_];

    for (int dt = 0; dt < 16; ++dt) {
      const int d0 = dt * 32;
      v8h xi_lo = *reinterpret_cast<const v8h*>(eirow + d0 + off);
      v8h xi_hi = *reinterpret_cast<const v8h*>(eirow + d0 + 16 + off);
      v8h xj_lo = *reinterpret_cast<const v8h*>(ejrow + d0 + off);
      v8h xj_hi = *reinterpret_cast<const v8h*>(ejrow + d0 + 16 + off);

      // a_c = ei*ej (packed f16 mul); a_d = |ei-ej| (packed sub + sign mask)
      v8h pl = xi_lo * xj_lo;
      v8h ph = xi_hi * xj_hi;
      v8h dl = xi_lo - xj_lo;
      v8h dh = xi_hi - xj_hi;
      v4u al = __builtin_bit_cast(v4u, dl) & 0x7FFF7FFFu;
      v4u ah = __builtin_bit_cast(v4u, dh) & 0x7FFF7FFFu;
      v16h a_c = __builtin_shufflevector(pl, ph, 0, 1, 2, 3, 4, 5, 6, 7,
                                         8, 9, 10, 11, 12, 13, 14, 15);
      v16h a_d = __builtin_shufflevector(__builtin_bit_cast(v8h, al),
                                         __builtin_bit_cast(v8h, ah),
                                         0, 1, 2, 3, 4, 5, 6, 7,
                                         8, 9, 10, 11, 12, 13, 14, 15);

      // B fragments: 4 k-tiles for this wave, pre-swizzled, 32B/lane each
      const size_t fb = ((size_t)dt * 32 + wave * 4) * 512 + (size_t)lane * 16;
      if (dt < 15) {  // prefetch next d-step's fragment stream toward the WGP
        __builtin_prefetch(wcf + fb + 32 * 512, 0, 3);
        __builtin_prefetch(wdf + fb + 32 * 512, 0, 3);
      }
#pragma unroll
      for (int t = 0; t < 4; ++t) {
        v16h bc = *reinterpret_cast<const v16h*>(wcf + fb + (size_t)t * 512);
        v16h bd = *reinterpret_cast<const v16h*>(wdf + fb + (size_t)t * 512);
        acc[t] = __builtin_amdgcn_wmma_f32_16x16x32_f16(false, a_c, false, bc,
                                                        (short)0, acc[t], false, false);
        acc[t] = __builtin_amdgcn_wmma_f32_16x16x32_f16(false, a_d, false, bd,
                                                        (short)0, acc[t], false, false);
      }
    }

    // fused epilogue: +a_i +c_j +b1 -> relu -> *W2, reduce over k via LDS.
    // Only 4 loads here (aproj row); everything else is in registers.
    const float* aprow = aprow_base + (size_t)il * D_;
    float ab[4];
#pragma unroll
    for (int t = 0; t < 4; ++t) ab[t] = aprow[kbase + t * 16] + b1k[t];
#pragma unroll
    for (int t = 0; t < 4; ++t) {
#pragma unroll
      for (int r = 0; r < 8; ++r) {
        const int jl = r + (lane >> 4) * 8;
        float v = acc[t][r] + ab[t] + cjk[t][r];
        v = fmaxf(v, 0.f) * w2k[t];
        atomicAdd(&s_scores[il * 16 + jl], v);
      }
    }
  }
  __syncthreads();

  {
    const int il = tid >> 4, jl = tid & 15;
    out[((size_t)b * N_ + (i0 + il)) * N_ + (j0 + jl)] = s_scores[tid] + b2[0];
  }
}

// ---------------------------------------------------------------------------
extern "C" void kernel_launch(void* const* d_in, const int* in_sizes, int n_in,
                              void* d_out, int out_size, void* d_ws, size_t ws_size,
                              hipStream_t stream) {
  const float* e  = (const float*)d_in[0];
  const float* W1 = (const float*)d_in[1];
  const float* b1 = (const float*)d_in[2];
  const float* W2 = (const float*)d_in[3];
  const float* b2 = (const float*)d_in[4];
  float* out = (float*)d_out;

  char* ws = (char*)d_ws;
  _Float16* eh    = (_Float16*)(ws + EH_OFF);
  float*    aproj = (float*)(ws + APROJ_OFF);
  float*    cproj = (float*)(ws + CPROJ_OFF);
  _Float16* wcf   = (_Float16*)(ws + WCF_OFF);
  _Float16* wdf   = (_Float16*)(ws + WDF_OFF);

  eh_kernel<<<(B_ * N_ * D_) / 256, 256, 0, stream>>>(e, eh);
  proj_kernel<<<(2 * B_ * N_ * D_) / 256, 256, 0, stream>>>(e, W1, aproj, cproj);
  pack_kernel<<<(2 * 16 * 32 * 512) / 256, 256, 0, stream>>>(W1, wcf, wdf);
  pairhead_main<<<B_ * 16 * 16, 256, 0, stream>>>(eh, aproj, cproj, wcf, wdf,
                                                  b1, W2, b2, out);
}
